// DualDiffusionEGNN_46076409151884
// MI455X (gfx1250) — matrix operand order repaired and checked by reading, not compiled, not run on hardware
//
#include <hip/hip_runtime.h>
#include <math.h>

// ---------------- problem constants ----------------
#define BATCH  4
#define NATOM  128
#define NDIM   128
#define EDIM   64
#define HDIM   256
#define NLAYER 6
#define NEDGE  (NATOM*(NATOM-1))   // 16256 edges per graph (multiple of 128)
#define EPN    (NATOM-1)           // 127 edges per node (contiguous by ROW)
#define KEF    352                 // 2*ND+1+ED = 321 padded to %32

// ---------------- CDNA5 WMMA types ----------------
typedef __attribute__((ext_vector_type(16))) __bf16 v16bf;
typedef __attribute__((ext_vector_type(8)))  float  v8f;

union __align__(16) FragBF { unsigned short u[16]; v16bf v; };

__device__ __forceinline__ unsigned short f2bf(float f) {
  unsigned int u = __float_as_uint(f);
  u = (u + 0x7FFFu + ((u >> 16) & 1u)) >> 16;   // RNE
  return (unsigned short)u;
}
__device__ __forceinline__ float bf2f(unsigned short h) {
  return __uint_as_float(((unsigned int)h) << 16);
}

// ============================================================
// bf16 WMMA GEMM:  D = act(A[M,K] @ W[K,Np] + bias)
// A bf16 row-major (lda). W passed COLUMN-MAJOR: Wt[n*K + k].
// act: 0=none 1=silu 2=tanh 3=sigmoid.  outbf: 1 -> bf16, 0 -> fp32.
//
// block = 128 threads = 4 waves. Block tile = 128(M) x 16*NT(N).
// Wave owns 2 M-tiles x NT N-tiles -> 2*NT WMMAs per 32-K step;
// A fragments reused across NT, B fragments across 2 M-tiles.
// The NT*16-column weight panel (contiguous in column-major W,
// <=24KB) is staged to LDS ONCE; steady-state loop is
// 4x global_load_b128 (A) + 2*NT ds_load_b128 (B) + 2*NT v_wmma.
// Requires M % 128 == 0, K % 32 == 0, Np % (16*NT) == 0.
// ============================================================
template <int NT>
__global__ void __launch_bounds__(128)
wmma_gemm_kernel(const unsigned short* __restrict__ A, int M, int lda,
                 const unsigned short* __restrict__ Wt, int K, int Np,
                 const float* __restrict__ bias,
                 void* __restrict__ Dv, int ldd, int act, int outbf) {
  extern __shared__ uint4 lds_raw[];                 // NT*16*K halves, 16B aligned
  unsigned short* lwT = (unsigned short*)lds_raw;

  const int lane = threadIdx.x & 31;
  const int wave = threadIdx.x >> 5;
  const int n0 = blockIdx.y * (16 * NT);
  const int m0 = blockIdx.x * 128 + wave * 32;       // two 16-row tiles per wave

  // ---- stage NT*16-column weight panel (contiguous in column-major W) ----
  {
    const uint4* src = (const uint4*)(Wt + (size_t)n0 * K);
    const int total = (16 * NT * K) >> 3;            // uint4 = 8 halves
    for (int i = threadIdx.x; i < total; i += 128) lds_raw[i] = src[i];
  }
  __syncthreads();

  v8f acc[2][NT];
#pragma unroll
  for (int t = 0; t < 2; ++t)
#pragma unroll
    for (int nt = 0; nt < NT; ++nt)
      acc[t][nt] = (v8f){0.f, 0.f, 0.f, 0.f, 0.f, 0.f, 0.f, 0.f};

  // A-fragment lane mapping (ISA 16-bit A 16x32): lane = m + 16*khalf
  const int am = lane & 15;
  const int kh = lane >> 4;          // 0: K 0..7 & 16..23 ; 1: K 8..15 & 24..31
  const unsigned short* arow0 = A + (size_t)(m0 + am) * lda + kh * 8;
  const unsigned short* arow1 = arow0 + (size_t)16 * lda;
  // B-fragment lane mapping: n = lane&15, K = (lane>>4)*8 + {0..7, 16..23} (matches A)
  const unsigned short* bcol[NT];
#pragma unroll
  for (int nt = 0; nt < NT; ++nt)
    bcol[nt] = lwT + (size_t)(nt * 16 + (lane & 15)) * K + (lane >> 4) * 8;

  for (int k0 = 0; k0 < K; k0 += 32) {
    FragBF a0, a1;
    *(uint4*)&a0.u[0] = *(const uint4*)(arow0 + k0);
    *(uint4*)&a0.u[8] = *(const uint4*)(arow0 + k0 + 16);
    *(uint4*)&a1.u[0] = *(const uint4*)(arow1 + k0);
    *(uint4*)&a1.u[8] = *(const uint4*)(arow1 + k0 + 16);
    if (k0 + 32 < K) {
      __builtin_prefetch(arow0 + k0 + 32, 0, 0);
      __builtin_prefetch(arow1 + k0 + 32, 0, 0);
    }
#pragma unroll
    for (int nt = 0; nt < NT; ++nt) {
      FragBF b;
      *(uint4*)&b.u[0] = *(const uint4*)(bcol[nt] + k0);       // ds_load_b128
      *(uint4*)&b.u[8] = *(const uint4*)(bcol[nt] + k0 + 16);  // ds_load_b128
      acc[0][nt] = __builtin_amdgcn_wmma_f32_16x16x32_bf16(
          false, a0.v, false, b.v, (short)0, acc[0][nt], false, false);
      acc[1][nt] = __builtin_amdgcn_wmma_f32_16x16x32_bf16(
          false, a1.v, false, b.v, (short)0, acc[1][nt], false, false);
    }
  }

  // C/D layout: lane n = lane&15; VGPR j -> row tile + j + (lane>>4)*8
  const int n = lane & 15;
  const int mo = (lane >> 4) * 8;
#pragma unroll
  for (int t = 0; t < 2; ++t) {
    const int rbase = m0 + t * 16 + mo;
#pragma unroll
    for (int nt = 0; nt < NT; ++nt) {
      const int col = n0 + nt * 16 + n;
      const float bv = bias ? bias[col] : 0.f;
#pragma unroll
      for (int j = 0; j < 8; ++j) {
        float v = acc[t][nt][j] + bv;
        if (act == 1)      v = v / (1.f + __expf(-v));       // silu
        else if (act == 2) v = tanhf(v);
        else if (act == 3) v = 1.f / (1.f + __expf(-v));     // sigmoid
        const size_t o = (size_t)(rbase + j) * ldd + col;
        if (outbf) ((unsigned short*)Dv)[o] = f2bf(v);
        else       ((float*)Dv)[o] = v;
      }
    }
  }
}

// ---------------- weight prep: fp32 [K,Nc] -> bf16 COLUMN-MAJOR [Np][Kp] ----------------
__global__ void pad_cast_kernel(const float* __restrict__ src,
                                unsigned short* __restrict__ dst,
                                int K, int Nc, int Kp, int Np) {
  int idx = blockIdx.x * 256 + threadIdx.x;     // idx = n*Kp + k
  if (idx >= Kp * Np) return;
  int n = idx / Kp, k = idx - n * Kp;
  dst[idx] = (k < K && n < Nc) ? f2bf(src[k * Nc + n]) : (unsigned short)0;
}

__global__ void pad_bias_kernel(const float* __restrict__ src,
                                float* __restrict__ dst, int n, int np) {
  int i = threadIdx.x;
  if (i < np) dst[i] = (i < n) ? src[i] : 0.f;
}

// ---------------- timestep + context embedding (tiny) ----------------
__global__ void __launch_bounds__(256)
time_ctx_kernel(const int* __restrict__ t, const float* __restrict__ pocket,
                const float* __restrict__ tw1, const float* __restrict__ tb1,
                const float* __restrict__ tw2, const float* __restrict__ tb2,
                const float* __restrict__ pw,  const float* __restrict__ pb,
                float* __restrict__ temb, float* __restrict__ ctx) {
  const int b = blockIdx.x, tid = threadIdx.x;
  __shared__ float emb[128];
  __shared__ float hid[256];
  const float tv = (float)t[b];
  if (tid < 64) {
    float fr = __expf(-9.210340371976184f * (float)tid / 64.f);
    float ang = tv * fr;
    emb[tid] = __sinf(ang);
    emb[tid + 64] = __cosf(ang);
  }
  __syncthreads();
  {
    float s = tb1[tid];
    for (int k = 0; k < 128; ++k) s += emb[k] * tw1[k * 256 + tid];
    hid[tid] = s / (1.f + __expf(-s));
  }
  __syncthreads();
  if (tid < 128) {
    float s = tb2[tid];
    for (int k = 0; k < 256; ++k) s += hid[k] * tw2[k * 128 + tid];
    temb[b * 128 + tid] = s;
    float s2 = pb[tid];
    for (int k = 0; k < 128; ++k) s2 += pocket[b * 128 + k] * pw[k * 128 + tid];
    ctx[b * 128 + tid] = s2;
  }
}

__global__ void h0_kernel(const int* __restrict__ atoms,
                          const float* __restrict__ atom_embed,
                          const float* __restrict__ temb,
                          const float* __restrict__ ctx,
                          float* __restrict__ h) {
  int idx = blockIdx.x * 256 + threadIdx.x;   // B*N*ND
  int b = idx >> 14, n = (idx >> 7) & 127, d = idx & 127;
  h[idx] = atom_embed[atoms[b * NATOM + n] * NDIM + d] +
           temb[b * 128 + d] + ctx[b * 128 + d];
}

__global__ void copy_x_kernel(const float* __restrict__ src,
                              float* __restrict__ dst, int n) {
  int i = blockIdx.x * 256 + threadIdx.x;
  if (i < n) dst[i] = src[i];
}

// ---------------- edge features (per graph, per layer) ----------------
// one wave per edge; lane writes 11 of the 352 padded ef columns
__global__ void __launch_bounds__(256)
edge_feat_kernel(const float* __restrict__ h, const float* __restrict__ x,
                 const int* __restrict__ bonds,
                 const float* __restrict__ bond_embed,
                 const float* __restrict__ dw, const float* __restrict__ db,
                 unsigned short* __restrict__ ef, float* __restrict__ unitv) {
  const int lane = threadIdx.x & 31;
  const int e = blockIdx.x * 8 + (threadIdx.x >> 5);
  const int r = e / EPN;
  const int j = e - r * EPN;
  const int cc = j + (j >= r ? 1 : 0);

  const float dx = x[r * 3 + 0] - x[cc * 3 + 0];
  const float dy = x[r * 3 + 1] - x[cc * 3 + 1];
  const float dz = x[r * 3 + 2] - x[cc * 3 + 2];
  const float dsq = dx * dx + dy * dy + dz * dz;
  const float dist = sqrtf(dsq);
  const int bond = bonds[r * NATOM + cc];

  if (lane == 0) {
    float inv = rsqrtf(dsq + 1e-8f);
    unitv[e * 3 + 0] = dx * inv;
    unitv[e * 3 + 1] = dy * inv;
    unitv[e * 3 + 2] = dz * inv;
  }
  unsigned short* dst = ef + (size_t)e * KEF;
#pragma unroll
  for (int ii = 0; ii < 11; ++ii) {
    const int i = lane * 11 + ii;
    float v;
    if (i < 128)        v = h[r * 128 + i];
    else if (i < 256)   v = h[cc * 128 + (i - 128)];
    else if (i == 256)  v = dsq;
    else if (i < 289)   v = bond_embed[bond * 32 + (i - 257)];
    else if (i < 321)   { int q = i - 289; v = dist * dw[q] + db[q]; }
    else                v = 0.f;
    dst[i] = f2bf(v);
  }
}

__global__ void scale_msg_kernel(unsigned short* __restrict__ msg,
                                 const float* __restrict__ att) {
  int idx = blockIdx.x * 256 + threadIdx.x;   // NEDGE*HDIM
  int e = idx >> 8;
  msg[idx] = f2bf(bf2f(msg[idx]) * att[(size_t)e * 16]);
}

// segment-sum over the 127 contiguous edges of each node; update x in place
__global__ void __launch_bounds__(256)
aggregate_update_kernel(const float* __restrict__ h,
                        const unsigned short* __restrict__ msg,
                        const float* __restrict__ coordw,
                        const float* __restrict__ unitv,
                        float* __restrict__ x,
                        unsigned short* __restrict__ node_in) {
  const int r = blockIdx.x, tid = threadIdx.x;
  const size_t e0 = (size_t)r * EPN;
  float s = 0.f;
  for (int j = 0; j < EPN; ++j) s += bf2f(msg[(e0 + j) * HDIM + tid]);
  node_in[r * 384 + 128 + tid] = f2bf(s);
  if (tid < 128) node_in[r * 384 + tid] = f2bf(h[r * 128 + tid]);

  __shared__ float red[3][128];
  if (tid < 128) {
    float w = 0.f, ux = 0.f, uy = 0.f, uz = 0.f;
    if (tid < EPN) {
      size_t e = e0 + tid;
      w = coordw[e * 16];
      ux = unitv[e * 3 + 0]; uy = unitv[e * 3 + 1]; uz = unitv[e * 3 + 2];
    }
    red[0][tid] = w * ux; red[1][tid] = w * uy; red[2][tid] = w * uz;
  }
  __syncthreads();
  for (int st = 64; st > 0; st >>= 1) {
    if (tid < st) {
      red[0][tid] += red[0][tid + st];
      red[1][tid] += red[1][tid + st];
      red[2][tid] += red[2][tid + st];
    }
    __syncthreads();
  }
  if (tid < 3) x[r * 3 + tid] += red[tid][0] / ((float)EPN + 1e-8f);
}

__global__ void __launch_bounds__(128)
ln_update_kernel(float* __restrict__ h, const float* __restrict__ hnew,
                 const float* __restrict__ g, const float* __restrict__ be) {
  const int r = blockIdx.x, d = threadIdx.x;
  __shared__ float sred[128];
  __shared__ float smu, svar;
  const float v = h[r * 128 + d] + hnew[r * 128 + d];
  sred[d] = v; __syncthreads();
  for (int st = 64; st > 0; st >>= 1) {
    if (d < st) sred[d] += sred[d + st];
    __syncthreads();
  }
  if (d == 0) smu = sred[0] / 128.f;
  __syncthreads();
  const float mu = smu, dv = v - mu;
  sred[d] = dv * dv; __syncthreads();
  for (int st = 64; st > 0; st >>= 1) {
    if (d < st) sred[d] += sred[d + st];
    __syncthreads();
  }
  if (d == 0) svar = sred[0] / 128.f;
  __syncthreads();
  h[r * 128 + d] = dv * rsqrtf(svar + 1e-5f) * g[d] + be[d];
}

__global__ void cast_h_kernel(const float* __restrict__ h,
                              unsigned short* __restrict__ hb) {
  int i = blockIdx.x * 256 + threadIdx.x;   // N*ND
  hb[i] = f2bf(h[i]);
}

__global__ void bond_in_kernel(const unsigned short* __restrict__ hb,
                               unsigned short* __restrict__ bi) {
  int idx = blockIdx.x * 256 + threadIdx.x;   // N*N*128
  int p = idx >> 7, d = idx & 127;
  int i = p >> 7, j = p & 127;
  bi[(size_t)p * 256 + d] = hb[i * 128 + d];
  bi[(size_t)p * 256 + 128 + d] = hb[j * 128 + d];
}

__global__ void head_store_kernel(const float* __restrict__ src,
                                  const float* __restrict__ mask,
                                  float* __restrict__ dst,
                                  int rows, int cols, int maskdiv) {
  int idx = blockIdx.x * 256 + threadIdx.x;
  if (idx >= rows * cols) return;
  int row = idx / cols, cc = idx - row * cols;
  dst[idx] = src[(size_t)row * 16 + cc] * mask[row / maskdiv];
}

// ============================================================
// host orchestration
// ============================================================
extern "C" void kernel_launch(void* const* d_in, const int* in_sizes, int n_in,
                              void* d_out, int out_size, void* d_ws, size_t ws_size,
                              hipStream_t stream) {
  if (n_in < 118) return;   // expects jax-pytree (sorted-key) flattening of params

  const float* xt     = (const float*)d_in[0];
  const int*   atoms  = (const int*)d_in[1];
  const int*   bonds  = (const int*)d_in[2];
  const int*   tt     = (const int*)d_in[3];
  const float* pocket = (const float*)d_in[4];
  const float* amask  = (const float*)d_in[5];

  // params (sorted): ah1 ah2 ahb1 ahb2 atom_embed bh1 bh2 bhb1 bhb2 bond_embed
  //                  ch1 ch2 chb1 chb2 db dw layers(6x15) pb pw tb1 tb2 tw1 tw2
  const float* ah1  = (const float*)d_in[6];
  const float* ah2  = (const float*)d_in[7];
  const float* ahb1 = (const float*)d_in[8];
  const float* ahb2 = (const float*)d_in[9];
  const float* atom_embed = (const float*)d_in[10];
  const float* bh1  = (const float*)d_in[11];
  const float* bh2  = (const float*)d_in[12];
  const float* bhb1 = (const float*)d_in[13];
  const float* bhb2 = (const float*)d_in[14];
  const float* bond_embed = (const float*)d_in[15];
  const float* ch1  = (const float*)d_in[16];
  const float* ch2  = (const float*)d_in[17];
  const float* chb1 = (const float*)d_in[18];
  const float* chb2 = (const float*)d_in[19];
  const float* dbp  = (const float*)d_in[20];
  const float* dwp  = (const float*)d_in[21];
  const int LB = 22;  // layer keys sorted: ab aw be cb1 cw1 cw2 eb1 eb2 ew1 ew2 g nb1 nb2 nw1 nw2
  enum { AB=0, AW, BE, CB1, CW1, CW2, EB1, EB2, EW1, EW2, G, NB1, NB2, NW1, NW2 };
  auto LP = [&](int l, int k) -> const float* {
    return (const float*)d_in[LB + 15 * l + k];
  };
  const float* pb  = (const float*)d_in[112];
  const float* pw  = (const float*)d_in[113];
  const float* tb1 = (const float*)d_in[114];
  const float* tb2 = (const float*)d_in[115];
  const float* tw1 = (const float*)d_in[116];
  const float* tw2 = (const float*)d_in[117];

  // -------- workspace carve --------
  char* wsp = (char*)d_ws;
  size_t off = 0;
  auto carve = [&](size_t bytes) -> void* {
    void* p = wsp + off;
    off += (bytes + 255) & ~(size_t)255;
    return p;
  };
  unsigned short *Wew1[NLAYER], *Wew2[NLAYER], *Waw[NLAYER], *Wcw1[NLAYER],
                 *Wcw2[NLAYER], *Wnw1[NLAYER], *Wnw2[NLAYER];
  float* Bab[NLAYER];
  for (int l = 0; l < NLAYER; ++l) {
    Wew1[l] = (unsigned short*)carve((size_t)KEF * 256 * 2);
    Wew2[l] = (unsigned short*)carve(256 * 256 * 2);
    Waw[l]  = (unsigned short*)carve(256 * 16 * 2);
    Wcw1[l] = (unsigned short*)carve(256 * 256 * 2);
    Wcw2[l] = (unsigned short*)carve(256 * 16 * 2);
    Wnw1[l] = (unsigned short*)carve(384 * 256 * 2);
    Wnw2[l] = (unsigned short*)carve(256 * 128 * 2);
    Bab[l]  = (float*)carve(16 * 4);
  }
  unsigned short* Wch1 = (unsigned short*)carve(128 * 256 * 2);
  unsigned short* Wch2 = (unsigned short*)carve(256 * 16 * 2);
  unsigned short* Wah1 = (unsigned short*)carve(128 * 256 * 2);
  unsigned short* Wah2 = (unsigned short*)carve(256 * 16 * 2);
  unsigned short* Wbh1 = (unsigned short*)carve(256 * 256 * 2);
  unsigned short* Wbh2 = (unsigned short*)carve(256 * 16 * 2);
  float* Bchb2 = (float*)carve(16 * 4);
  float* Bahb2 = (float*)carve(16 * 4);
  float* Bbhb2 = (float*)carve(16 * 4);

  float* temb = (float*)carve(BATCH * 128 * 4);
  float* ctxv = (float*)carve(BATCH * 128 * 4);
  float* hbuf = (float*)carve((size_t)BATCH * NATOM * NDIM * 4);
  float* xbuf = (float*)carve((size_t)BATCH * NATOM * 3 * 4);
  unsigned short* ef  = (unsigned short*)carve((size_t)NEDGE * KEF * 2);
  unsigned short* y1  = (unsigned short*)carve((size_t)NEDGE * HDIM * 2);
  unsigned short* msg = (unsigned short*)carve((size_t)NEDGE * HDIM * 2);
  unsigned short* chh = (unsigned short*)carve((size_t)NEDGE * HDIM * 2);
  float* att   = (float*)carve((size_t)NEDGE * 16 * 4);
  float* cwv   = (float*)carve((size_t)NEDGE * 16 * 4);
  float* unitv = (float*)carve((size_t)NEDGE * 3 * 4);
  unsigned short* node_in = (unsigned short*)carve(NATOM * 384 * 2);
  unsigned short* n1      = (unsigned short*)carve(NATOM * 256 * 2);
  float* hnew             = (float*)carve(NATOM * 128 * 4);
  unsigned short* hb16    = (unsigned short*)carve(NATOM * 128 * 2);
  unsigned short* t1      = (unsigned short*)carve(NATOM * 256 * 2);
  unsigned short* bondi   = (unsigned short*)carve((size_t)NATOM * NATOM * 256 * 2);
  unsigned short* bondh   = (unsigned short*)carve((size_t)NATOM * NATOM * 256 * 2);
  float* head16           = (float*)carve((size_t)NATOM * NATOM * 16 * 4);
  (void)ws_size; (void)in_sizes; (void)out_size;

  auto padcast = [&](const float* s, unsigned short* d, int K, int Nc, int Kp, int Np) {
    int tot = Kp * Np;
    pad_cast_kernel<<<(tot + 255) / 256, 256, 0, stream>>>(s, d, K, Nc, Kp, Np);
  };
  auto padbias = [&](const float* s, float* d, int n, int np) {
    pad_bias_kernel<<<1, np, 0, stream>>>(s, d, n, np);
  };
  auto gemm = [&](const unsigned short* A, int M, int lda,
                  const unsigned short* Wt, int K, int Np,
                  const float* bias, void* D, int ldd, int act, int outbf) {
    if (Np % 32 == 0) {
      dim3 g(M / 128, Np / 32);
      size_t shmem = (size_t)K * 32 * 2;   // 32-column weight panel
      wmma_gemm_kernel<2><<<g, 128, shmem, stream>>>(A, M, lda, Wt, K, Np, bias,
                                                     D, ldd, act, outbf);
    } else {
      dim3 g(M / 128, Np / 16);
      size_t shmem = (size_t)K * 16 * 2;   // 16-column weight panel
      wmma_gemm_kernel<1><<<g, 128, shmem, stream>>>(A, M, lda, Wt, K, Np, bias,
                                                     D, ldd, act, outbf);
    }
  };

  // -------- weight conversion (deterministic, every call) --------
  for (int l = 0; l < NLAYER; ++l) {
    padcast(LP(l, EW1), Wew1[l], 321, 256, KEF, 256);
    padcast(LP(l, EW2), Wew2[l], 256, 256, 256, 256);
    padcast(LP(l, AW),  Waw[l],  256, 1,   256, 16);
    padcast(LP(l, CW1), Wcw1[l], 256, 256, 256, 256);
    padcast(LP(l, CW2), Wcw2[l], 256, 1,   256, 16);
    padcast(LP(l, NW1), Wnw1[l], 384, 256, 384, 256);
    padcast(LP(l, NW2), Wnw2[l], 256, 128, 256, 128);
    padbias(LP(l, AB), Bab[l], 1, 16);
  }
  padcast(ch1, Wch1, 128, 256, 128, 256);
  padcast(ch2, Wch2, 256, 3,   256, 16);
  padbias(chb2, Bchb2, 3, 16);
  padcast(ah1, Wah1, 128, 256, 128, 256);
  padcast(ah2, Wah2, 256, 11,  256, 16);
  padbias(ahb2, Bahb2, 11, 16);
  padcast(bh1, Wbh1, 256, 256, 256, 256);
  padcast(bh2, Wbh2, 256, 5,   256, 16);
  padbias(bhb2, Bbhb2, 5, 16);

  // -------- embeddings --------
  time_ctx_kernel<<<BATCH, 256, 0, stream>>>(tt, pocket, tw1, tb1, tw2, tb2,
                                             pw, pb, temb, ctxv);
  h0_kernel<<<(BATCH * NATOM * NDIM) / 256, 256, 0, stream>>>(
      atoms, atom_embed, temb, ctxv, hbuf);
  copy_x_kernel<<<(BATCH * NATOM * 3 + 255) / 256, 256, 0, stream>>>(
      xt, xbuf, BATCH * NATOM * 3);

  float* out = (float*)d_out;
  float* out_noise = out;
  float* out_atom  = out + BATCH * NATOM * 3;
  float* out_bond  = out + BATCH * NATOM * 3 + BATCH * NATOM * 11;

  for (int b = 0; b < BATCH; ++b) {
    float* hB = hbuf + (size_t)b * NATOM * NDIM;
    float* xB = xbuf + (size_t)b * NATOM * 3;
    const int* bondsB = bonds + (size_t)b * NATOM * NATOM;
    const float* maskB = amask + (size_t)b * NATOM;

    for (int l = 0; l < NLAYER; ++l) {
      edge_feat_kernel<<<NEDGE / 8, 256, 0, stream>>>(
          hB, xB, bondsB, bond_embed, dwp, dbp, ef, unitv);
      // edge MLP + attention
      gemm(ef,  NEDGE, KEF, Wew1[l], KEF, 256, LP(l, EB1), y1,  256, 1, 1);
      gemm(y1,  NEDGE, 256, Wew2[l], 256, 256, LP(l, EB2), msg, 256, 1, 1);
      gemm(msg, NEDGE, 256, Waw[l],  256, 16,  Bab[l],     att, 16,  3, 0);
      scale_msg_kernel<<<(NEDGE * HDIM) / 256, 256, 0, stream>>>(msg, att);
      // coord MLP
      gemm(msg, NEDGE, 256, Wcw1[l], 256, 256, LP(l, CB1), chh, 256, 1, 1);
      gemm(chh, NEDGE, 256, Wcw2[l], 256, 16,  nullptr,    cwv, 16,  2, 0);
      // aggregate msg + coord update, build node-MLP input
      aggregate_update_kernel<<<NATOM, 256, 0, stream>>>(
          hB, msg, cwv, unitv, xB, node_in);
      // node MLP + residual LN
      gemm(node_in, NATOM, 384, Wnw1[l], 384, 256, LP(l, NB1), n1,   256, 1, 1);
      gemm(n1,      NATOM, 256, Wnw2[l], 256, 128, LP(l, NB2), hnew, 128, 0, 0);
      ln_update_kernel<<<NATOM, 128, 0, stream>>>(hB, hnew, LP(l, G), LP(l, BE));
    }

    cast_h_kernel<<<(NATOM * NDIM) / 256, 256, 0, stream>>>(hB, hb16);

    // noise head
    gemm(hb16, NATOM, 128, Wch1, 128, 256, chb1,  t1,     256, 1, 1);
    gemm(t1,   NATOM, 256, Wch2, 256, 16,  Bchb2, head16, 16,  0, 0);
    head_store_kernel<<<(NATOM * 3 + 255) / 256, 256, 0, stream>>>(
        head16, maskB, out_noise + (size_t)b * NATOM * 3, NATOM, 3, 1);
    // atom head
    gemm(hb16, NATOM, 128, Wah1, 128, 256, ahb1,  t1,     256, 1, 1);
    gemm(t1,   NATOM, 256, Wah2, 256, 16,  Bahb2, head16, 16,  0, 0);
    head_store_kernel<<<(NATOM * 11 + 255) / 256, 256, 0, stream>>>(
        head16, maskB, out_atom + (size_t)b * NATOM * 11, NATOM, 11, 1);
    // bond head (N*N rows)
    bond_in_kernel<<<(NATOM * NATOM * 128) / 256, 256, 0, stream>>>(hb16, bondi);
    gemm(bondi, NATOM * NATOM, 256, Wbh1, 256, 256, bhb1,  bondh,  256, 1, 1);
    gemm(bondh, NATOM * NATOM, 256, Wbh2, 256, 16,  Bbhb2, head16, 16,  0, 0);
    head_store_kernel<<<(NATOM * NATOM * 5 + 255) / 256, 256, 0, stream>>>(
        head16, maskB, out_bond + (size_t)b * NATOM * NATOM * 5,
        NATOM * NATOM, 5, NATOM);
  }
}
